// WindowAttention_64467459113142
// MI455X (gfx1250) — matrix operand order repaired and checked by reading, not compiled, not run on hardware
//
#include <hip/hip_runtime.h>

// ---------------- problem constants ----------------
#define DIMC   192
#define NTOK   49
#define NPAD   64
#define HEADS  6
#define HD     32
#define C3     576
#define NWIN   8192
#define NWMASK 1024
#define SCALE  0.17677669529663687f   // 1/sqrt(32)
#define NEGBIG (-1.0e30f)

typedef __attribute__((ext_vector_type(16))) _Float16 v16h;
typedef __attribute__((ext_vector_type(8)))  _Float16 v8h;
typedef __attribute__((ext_vector_type(8)))  float    v8f;

// ---------------- LDS layout (bytes / f16-element strides) ----------------
#define XA_OFF     0        // [64][200] f16 : x(f16) during QKV, attn*v later
#define XA_STR     200
#define QKH_OFF    25600    // [64][392] f16 : cols 0..191 = q*scale, 192..383 = k
#define QKH_STR    392
#define VT_OFF     75776    // [192][72] f16 : v transposed (d-major)
#define VT_STR     72
#define AF_OFF     103424   // [64][64] f32 : attention logits
#define AH_OFF     119808   // [64][64] f16 : softmax probs
#define SMEM_BYTES 128000

// ---------------- workspace layout (bytes) ----------------
#define WS_QKVWH   0                       // 576*192 f16
#define WS_PROJWH  221184                  // 192*192 f16
#define WS_BIAS64  294912                  // 6*64*64 f32 (padded with zeros)
#define WS_BYTES   393216

union AFrag { v16h v; v8h h[2]; };

// A-matrix fragment (16x32 f16): lane half (lane>>4) selects K sub-runs.
// lanes 0-15: K = kbase+0..15 ; lanes 16-31: K = kbase+16..31.
__device__ inline v16h load_a_frag(const _Float16* rowp, int kbase, int half) {
    AFrag a;
    a.h[0] = *(const v8h*)(rowp + kbase + half * 16);
    a.h[1] = *(const v8h*)(rowp + kbase + half * 16 + 8);
    return a.v;
}

// ---------------- pre-pass: weights f32 -> f16 ----------------
__global__ void prep_weights(const float* __restrict__ qkv_w,
                             const float* __restrict__ proj_w,
                             _Float16* __restrict__ qkv_w_h,
                             _Float16* __restrict__ proj_w_h) {
    int i = blockIdx.x * 256 + threadIdx.x;
    if (i < C3 * DIMC)   qkv_w_h[i]  = (_Float16)qkv_w[i];
    if (i < DIMC * DIMC) proj_w_h[i] = (_Float16)proj_w[i];
}

// ---------------- pre-pass: padded relative-position bias [6][64][64] ----------------
__global__ void prep_bias(const float* __restrict__ table,   // [169][6]
                          const int* __restrict__ rel_index, // [49][49]
                          float* __restrict__ bias64) {
    int i = blockIdx.x * 256 + threadIdx.x;      // 6*64*64 = 24576
    if (i >= HEADS * NPAD * NPAD) return;
    int h = i >> 12;
    int r = (i >> 6) & 63;
    int c = i & 63;
    float v = 0.0f;
    if (r < NTOK && c < NTOK) v = table[rel_index[r * NTOK + c] * HEADS + h];
    bias64[i] = v;
}

// ---------------- fused window attention: one block per window ----------------
__global__ void __launch_bounds__(256)
win_attn(const float* __restrict__ x,        // [8192][49][192]
         const float* __restrict__ mask,     // [1024][49][49]
         const float* __restrict__ qkv_b,    // [576]
         const float* __restrict__ proj_b,   // [192]
         const _Float16* __restrict__ qkvWh, // [576][192] f16
         const _Float16* __restrict__ projWh,// [192][192] f16
         const float* __restrict__ bias64,   // [6][64][64]
         float* __restrict__ out) {          // [8192][49][192]
    extern __shared__ __align__(16) char smem[];
    _Float16* XA  = (_Float16*)(smem + XA_OFF);
    _Float16* QKH = (_Float16*)(smem + QKH_OFF);
    _Float16* VT  = (_Float16*)(smem + VT_OFF);
    float*    AF  = (float*)(smem + AF_OFF);
    _Float16* AH  = (_Float16*)(smem + AH_OFF);

    const int tid  = threadIdx.x;
    const int wave = tid >> 5;
    const int lane = tid & 31;
    const int ln   = lane & 15;
    const int half = lane >> 4;
    const int w    = blockIdx.x;
    const int widx = w & (NWMASK - 1);

    // ---- phase 0: stage x window into LDS as f16, zero-pad rows 49..63 ----
    {
        const float* xw = x + (size_t)w * NTOK * DIMC;
        for (int i = tid; i < NPAD * DIMC; i += 256) {
            int r = i / DIMC, c = i - r * DIMC;
            float v = (r < NTOK) ? xw[r * DIMC + c] : 0.0f;
            XA[r * XA_STR + c] = (_Float16)v;
        }
    }
    __syncthreads();

    // ---- phase 1: QKV GEMM  [64x192] @ [192x576] ; fold bias + q-scale ----
    // nt is wave-uniform: 0..11 -> q (scaled), 12..23 -> k, 24..35 -> v (transposed store)
    for (int nt = wave; nt < 36; nt += 8) {
        const int ncol = nt * 16 + ln;
        v16h Bf[6];
#pragma unroll
        for (int kk = 0; kk < 6; kk++)
            Bf[kk] = *(const v16h*)(qkvWh + (size_t)ncol * DIMC + kk * 32 + half * 16);
        const float bias = qkv_b[ncol];
        const float mul  = (nt < 12) ? SCALE : 1.0f;   // uniform select
#pragma unroll
        for (int m = 0; m < 4; m++) {
            v8f acc = {};
            const _Float16* rowp = XA + (m * 16 + ln) * XA_STR;
#pragma unroll
            for (int kk = 0; kk < 6; kk++) {
                v16h a = load_a_frag(rowp, kk * 32, half);
                acc = __builtin_amdgcn_wmma_f32_16x16x32_f16(
                    false, a, false, Bf[kk], (short)0, acc, false, false);
            }
            if (nt < 24) {               // wave-uniform branch: q|k section
#pragma unroll
                for (int j = 0; j < 8; j++) {
                    int row = m * 16 + j + 8 * half;
                    QKH[row * QKH_STR + ncol] = (_Float16)((acc[j] + bias) * mul);
                }
            } else {                     // wave-uniform branch: v section, transposed
#pragma unroll
                for (int j = 0; j < 8; j++) {
                    int row = m * 16 + j + 8 * half;
                    VT[(ncol - 384) * VT_STR + row] = (_Float16)(acc[j] + bias);
                }
            }
        }
    }
    __syncthreads();

    // ---- phase 2: per-head attention ----
    for (int h = 0; h < HEADS; h++) {
        // (a) logits = (q*scale) @ k^T : 16 tiles of 16x16, K = hd = 32
#pragma unroll
        for (int i = 0; i < 2; i++) {
            int t = wave * 2 + i;          // 0..15
            int mt = t >> 2, nt2 = t & 3;
            const _Float16* qrow = QKH + (mt * 16 + ln) * QKH_STR + h * HD;
            v16h a = load_a_frag(qrow, 0, half);
            v16h b = *(const v16h*)(QKH + (nt2 * 16 + ln) * QKH_STR + DIMC + h * HD + half * 16);
            v8f acc = {};
            acc = __builtin_amdgcn_wmma_f32_16x16x32_f16(
                false, a, false, b, (short)0, acc, false, false);
#pragma unroll
            for (int j = 0; j < 8; j++)
                AF[(mt * 16 + j + 8 * half) * NPAD + nt2 * 16 + ln] = acc[j];
        }
        __syncthreads();

        // (b) softmax over columns (valid cols < 49), + rel-bias + mask
        {
            int row = tid >> 2;            // 0..63, 4 threads per row
            int q4  = tid & 3;             // each covers 16 columns
            const float* afr = AF + row * NPAD;
            const float* bh  = bias64 + (h * NPAD + row) * NPAD;
            const float* mr  = mask + (size_t)widx * (NTOK * NTOK) + row * NTOK;
            float vals[16];
            float mx = NEGBIG;
#pragma unroll
            for (int cc = 0; cc < 16; cc++) {
                int c = q4 * 16 + cc;
                float v = NEGBIG;
                if (row < NTOK && c < NTOK) v = afr[c] + bh[c] + mr[c];
                vals[cc] = v;
                mx = fmaxf(mx, v);
            }
            mx = fmaxf(mx, __shfl_xor(mx, 1));
            mx = fmaxf(mx, __shfl_xor(mx, 2));
            float s = 0.0f;
#pragma unroll
            for (int cc = 0; cc < 16; cc++) {
                float e = __expf(vals[cc] - mx);
                vals[cc] = e;
                s += e;
            }
            s += __shfl_xor(s, 1);
            s += __shfl_xor(s, 2);
            float inv = (row < NTOK) ? (1.0f / s) : 0.0f;  // pad rows -> all-zero probs
#pragma unroll
            for (int cc = 0; cc < 16; cc++)
                AH[row * NPAD + q4 * 16 + cc] = (_Float16)(vals[cc] * inv);
        }
        __syncthreads();

        // (c) out_h = probs @ v : 8 tiles (4 M x 2 hd), K = 64 positions
        {
            int mt = wave >> 1, nt2 = wave & 1;
            v8f acc = {};
            const _Float16* arow = AH + (mt * 16 + ln) * NPAD;
            const _Float16* vrow = VT + (h * HD + nt2 * 16 + ln) * VT_STR;
#pragma unroll
            for (int kk = 0; kk < 2; kk++) {
                v16h a = load_a_frag(arow, kk * 32, half);
                v16h b = *(const v16h*)(vrow + kk * 32 + half * 16);
                acc = __builtin_amdgcn_wmma_f32_16x16x32_f16(
                    false, a, false, b, (short)0, acc, false, false);
            }
#pragma unroll
            for (int j = 0; j < 8; j++)
                XA[(mt * 16 + j + 8 * half) * XA_STR + h * HD + nt2 * 16 + ln] = (_Float16)acc[j];
        }
        __syncthreads();
    }

    // ---- phase 3: proj GEMM  [64x192] @ [192x192] + bias, store 49 rows ----
    for (int nt = wave; nt < 12; nt += 8) {
        const int ncol = nt * 16 + ln;
        v16h Bf[6];
#pragma unroll
        for (int kk = 0; kk < 6; kk++)
            Bf[kk] = *(const v16h*)(projWh + (size_t)ncol * DIMC + kk * 32 + half * 16);
        const float pb = proj_b[ncol];
#pragma unroll
        for (int m = 0; m < 4; m++) {
            v8f acc = {};
            const _Float16* rowp = XA + (m * 16 + ln) * XA_STR;
#pragma unroll
            for (int kk = 0; kk < 6; kk++) {
                v16h a = load_a_frag(rowp, kk * 32, half);
                acc = __builtin_amdgcn_wmma_f32_16x16x32_f16(
                    false, a, false, Bf[kk], (short)0, acc, false, false);
            }
#pragma unroll
            for (int j = 0; j < 8; j++) {
                int row = m * 16 + j + 8 * half;
                if (row < NTOK)
                    out[(size_t)w * NTOK * DIMC + row * DIMC + ncol] = acc[j] + pb;
            }
        }
    }
}

extern "C" void kernel_launch(void* const* d_in, const int* in_sizes, int n_in,
                              void* d_out, int out_size, void* d_ws, size_t ws_size,
                              hipStream_t stream) {
    (void)in_sizes; (void)n_in; (void)out_size; (void)ws_size;
    const float* x      = (const float*)d_in[0];
    const float* mask   = (const float*)d_in[1];
    const float* qkv_w  = (const float*)d_in[2];
    const float* qkv_b  = (const float*)d_in[3];
    const float* proj_w = (const float*)d_in[4];
    const float* proj_b = (const float*)d_in[5];
    const float* rbt    = (const float*)d_in[6];
    const int*   ridx   = (const int*)d_in[7];

    char* ws = (char*)d_ws;
    _Float16* qkvWh  = (_Float16*)(ws + WS_QKVWH);
    _Float16* projWh = (_Float16*)(ws + WS_PROJWH);
    float*    bias64 = (float*)(ws + WS_BIAS64);

    prep_weights<<<(C3 * DIMC + 255) / 256, 256, 0, stream>>>(qkv_w, proj_w, qkvWh, projWh);
    prep_bias<<<(HEADS * NPAD * NPAD + 255) / 256, 256, 0, stream>>>(rbt, ridx, bias64);
    win_attn<<<NWIN, 256, SMEM_BYTES, stream>>>(x, mask, qkv_b, proj_b,
                                                qkvWh, projWh, bias64, (float*)d_out);
}
